// BitLinear158_20349555048665
// MI455X (gfx1250) — matrix operand order repaired and checked by reading, not compiled, not run on hardware
//
#include <hip/hip_runtime.h>
#include <stdint.h>

// ---------------------------------------------------------------------------
// BitLinear 1.58b:  out = x @ ternary(w).T + bias
//   x: [4,2048,4096] f32   w: [16384,4096] f32   bias: [16384] f32
//   M=8192, K=4096, N=16384  -> 1.1 TFLOP, compute-bound -> WMMA bf16 path,
//   async global->LDS double-buffered software pipeline.
// ---------------------------------------------------------------------------

#define IN_F   4096
#define OUT_F  16384
#define M_TOT  8192

typedef __attribute__((ext_vector_type(16))) __bf16 v16bf;
typedef __attribute__((ext_vector_type(8)))  float  v8f;

union FragU { uint4 u[2]; v16bf v; };

static __device__ __forceinline__ unsigned short f32_to_bf16_rne(float f) {
    union { float f; uint32_t u; } a; a.f = f;
    uint32_t u = a.u;
    uint32_t r = (u + 0x7FFFu + ((u >> 16) & 1u)) >> 16;  // round-to-nearest-even
    return (unsigned short)r;
}

// ------------------------- scale = mean(|w|) -------------------------------
__global__ void absum_partial(const float* __restrict__ w, float* __restrict__ part, int n) {
    __shared__ float sm[256];
    float s = 0.f;
    for (int i = blockIdx.x * blockDim.x + threadIdx.x; i < n; i += gridDim.x * blockDim.x)
        s += fabsf(w[i]);
    sm[threadIdx.x] = s;
    __syncthreads();
    for (int off = 128; off > 0; off >>= 1) {
        if ((int)threadIdx.x < off) sm[threadIdx.x] += sm[threadIdx.x + off];
        __syncthreads();
    }
    if (threadIdx.x == 0) part[blockIdx.x] = sm[0];
}

__global__ void absum_final(const float* __restrict__ part, float* __restrict__ thresh,
                            int npart, float inv_count) {
    __shared__ float sm[256];
    float s = 0.f;
    for (int i = threadIdx.x; i < npart; i += 256) s += part[i];
    sm[threadIdx.x] = s;
    __syncthreads();
    for (int off = 128; off > 0; off >>= 1) {
        if ((int)threadIdx.x < off) sm[threadIdx.x] += sm[threadIdx.x + off];
        __syncthreads();
    }
    if (threadIdx.x == 0) thresh[0] = 0.7f * sm[0] * inv_count;   // 0.7*mean(|w|)
}

// ---------------- ternarize weights -> bf16 {-1,0,+1} ----------------------
__global__ void quant_w_kernel(const float4* __restrict__ w, const float* __restrict__ thr,
                               ushort4* __restrict__ wq, int n4) {
    const float t = thr[0];
    for (int i = blockIdx.x * blockDim.x + threadIdx.x; i < n4; i += gridDim.x * blockDim.x) {
        float4 v = w[i];
        ushort4 o;
        o.x = (fabsf(v.x) >= t) ? (v.x >= 0.f ? 0x3F80u : 0xBF80u) : 0u;
        o.y = (fabsf(v.y) >= t) ? (v.y >= 0.f ? 0x3F80u : 0xBF80u) : 0u;
        o.z = (fabsf(v.z) >= t) ? (v.z >= 0.f ? 0x3F80u : 0xBF80u) : 0u;
        o.w = (fabsf(v.w) >= t) ? (v.w >= 0.f ? 0x3F80u : 0xBF80u) : 0u;
        wq[i] = o;
    }
}

// ---------------------- x fp32 -> bf16 -------------------------------------
__global__ void cvt_x_kernel(const float4* __restrict__ x, ushort4* __restrict__ xb, int n4) {
    for (int i = blockIdx.x * blockDim.x + threadIdx.x; i < n4; i += gridDim.x * blockDim.x) {
        float4 v = x[i];
        ushort4 o;
        o.x = f32_to_bf16_rne(v.x);
        o.y = f32_to_bf16_rne(v.y);
        o.z = f32_to_bf16_rne(v.z);
        o.w = f32_to_bf16_rne(v.w);
        xb[i] = o;
    }
}

// ------------------------- WMMA GEMM ---------------------------------------
// Block tile 128(M) x 128(N), K-step 64.  8 waves (wave32); each wave owns a
// 64x32 subtile = 4x2 grid of 16x16 WMMA accumulators.
// Double-buffered LDS fed by GLOBAL_LOAD_ASYNC_TO_LDS_B128 (ASYNCcnt).
#define BM 128
#define BN 128
#define BK 64
#define LDSW 72                   // bf16 per LDS row: 64 + 8 pad -> 144B rows
#define TILE_SH   (BM * LDSW)     // ushorts per tile (A or B): 9216
#define TILE_BY   (TILE_SH * 2)   // bytes per tile: 18432
#define BUF_BY    (2 * TILE_BY)   // bytes per buffer (A+B): 36864

__global__ __launch_bounds__(256)
void bitlinear_wmma(const unsigned short* __restrict__ A,   // x  bf16 [M,K]
                    const unsigned short* __restrict__ B,   // wq bf16 [N,K]
                    const float* __restrict__ bias,
                    float* __restrict__ C,                  // [M,N] f32
                    int M, int N, int K) {
    __shared__ __align__(16) unsigned short lds[2 * 2 * TILE_SH];

    // ptrtoint of the shared array: (a) escapes `lds` so the compiler cannot
    // fold the ds_loads of an "apparently never-written" LDS object to undef
    // (the only writers are the async-load asm below), and (b) flat->LDS
    // truncation means addr[31:0] IS the wave-relative LDS byte address the
    // async instruction's VDST operand expects.
    const unsigned ldsBase = (unsigned)(uintptr_t)(void*)lds;

    const int tid    = threadIdx.x;
    const int lane   = tid & 31;
    const int wave   = tid >> 5;
    const int lane16 = lane & 15;
    const int laneHi = lane >> 4;          // 0: lanes 0-15, 1: lanes 16-31

    const int m0 = blockIdx.y * BM;
    const int n0 = blockIdx.x * BN;
    const int wm = (wave >> 2) * 64;       // wave M offset in tile: 0 / 64
    const int wn = (wave & 3) * 32;        // wave N offset in tile: 0/32/64/96

    // Staging: tile = 128 rows x 64 bf16 = 1024 x 16B chunks; 4 per thread.
    // chunk i: row = tid/8 + 32*i, col = (tid%8)*8 bf16.
    const int r0  = tid >> 3;
    const int col = (tid & 7) * 8;
    unsigned lA_ad[4], lB_ad[4];     // LDS byte addresses (buffer 0)
    unsigned gA_off[4], gB_off[4];   // global byte offsets at kb=0
#pragma unroll
    for (int i = 0; i < 4; ++i) {
        const int row = r0 + 32 * i;
        lA_ad[i] = ldsBase + (unsigned)(row * LDSW + col) * 2u;
        lB_ad[i] = lA_ad[i] + (unsigned)TILE_BY;
        gA_off[i] = ((unsigned)(m0 + row) * (unsigned)K + (unsigned)col) * 2u;
        gB_off[i] = ((unsigned)(n0 + row) * (unsigned)K + (unsigned)col) * 2u;
    }

    const v8f vzero = {0.f, 0.f, 0.f, 0.f, 0.f, 0.f, 0.f, 0.f};
    v8f acc[4][2];
#pragma unroll
    for (int mt = 0; mt < 4; ++mt)
#pragma unroll
        for (int nt = 0; nt < 2; ++nt) acc[mt][nt] = vzero;

    // ---- prologue: async-stage tile kb=0 into buffer 0 ----
#pragma unroll
    for (int i = 0; i < 4; ++i) {
        asm volatile("global_load_async_to_lds_b128 %0, %1, %2"
                     :: "v"(lA_ad[i]), "v"(gA_off[i]), "s"(A) : "memory");
        asm volatile("global_load_async_to_lds_b128 %0, %1, %2"
                     :: "v"(lB_ad[i]), "v"(gB_off[i]), "s"(B) : "memory");
    }
    asm volatile("s_wait_asynccnt 0x0" ::: "memory");
    __syncthreads();

    unsigned buf = 0;
    for (int kb = 0; kb < K; kb += BK) {
        const bool more = (kb + BK) < K;
        // ---- async-stage next tile into the other buffer (overlaps WMMA) ----
        if (more) {
            const unsigned nb = (buf ^ 1u) * (unsigned)BUF_BY;
            const unsigned kadv = (unsigned)(kb + BK) * 2u;
#pragma unroll
            for (int i = 0; i < 4; ++i) {
                asm volatile("global_load_async_to_lds_b128 %0, %1, %2"
                             :: "v"(lA_ad[i] + nb), "v"(gA_off[i] + kadv), "s"(A)
                             : "memory");
                asm volatile("global_load_async_to_lds_b128 %0, %1, %2"
                             :: "v"(lB_ad[i] + nb), "v"(gB_off[i] + kadv), "s"(B)
                             : "memory");
            }
        }

        // ---- compute from current buffer ----
        const unsigned short* lA = lds + buf * (2 * TILE_SH);
        const unsigned short* lB = lA + TILE_SH;
#pragma unroll
        for (int ks = 0; ks < BK; ks += 32) {
            // B fragment (32x16): lane holds column n=lane16;
            // lanes 0-15: K = ks+{0..7,16..23}; lanes 16-31: +8.
            v16bf bfr[2];
#pragma unroll
            for (int nt = 0; nt < 2; ++nt) {
                const unsigned short* p =
                    lB + (wn + nt * 16 + lane16) * LDSW + ks + laneHi * 8;
                FragU f;
                f.u[0] = *(const uint4*)(p);
                f.u[1] = *(const uint4*)(p + 16);
                bfr[nt] = f.v;
            }
            // A fragment (16x32): lane holds row m=lane16;
            // lanes 0-15: K = ks+{0..7,16..23}; lanes 16-31: +8.
            v16bf afr[4];
#pragma unroll
            for (int mt = 0; mt < 4; ++mt) {
                const unsigned short* p =
                    lA + (wm + mt * 16 + lane16) * LDSW + ks + laneHi * 8;
                FragU f;
                f.u[0] = *(const uint4*)(p);
                f.u[1] = *(const uint4*)(p + 16);
                afr[mt] = f.v;
            }
#pragma unroll
            for (int mt = 0; mt < 4; ++mt)
#pragma unroll
                for (int nt = 0; nt < 2; ++nt)
                    acc[mt][nt] = __builtin_amdgcn_wmma_f32_16x16x32_bf16(
                        false, afr[mt], false, bfr[nt],
                        (short)0, acc[mt][nt], false, false);
        }

        // ---- publish next buffer ----
        if (more) asm volatile("s_wait_asynccnt 0x0" ::: "memory");
        __syncthreads();
        buf ^= 1u;
    }

    // Epilogue (D layout): lanes 0-15: VGPR r = (M=r, N=lane16);
    //                      lanes 16-31: VGPR r = (M=8+r, N=lane16).
#pragma unroll
    for (int nt = 0; nt < 2; ++nt) {
        const int n   = n0 + wn + nt * 16 + lane16;
        const float bv = bias[n];
#pragma unroll
        for (int mt = 0; mt < 4; ++mt) {
            const int mbase = m0 + wm + mt * 16 + laneHi * 8;
#pragma unroll
            for (int r = 0; r < 8; ++r)
                C[(size_t)(mbase + r) * N + n] = acc[mt][nt][r] + bv;
        }
    }
}

// ---------------------------------------------------------------------------
extern "C" void kernel_launch(void* const* d_in, const int* in_sizes, int n_in,
                              void* d_out, int out_size, void* d_ws, size_t ws_size,
                              hipStream_t stream) {
    const float* x    = (const float*)d_in[0];   // [4,2048,4096]
    const float* w    = (const float*)d_in[1];   // [16384,4096]
    const float* bias = (const float*)d_in[2];   // [16384]
    float* out = (float*)d_out;                  // [4,2048,16384]

    // Workspace: [0,4K) partials, [4K,4K+4) threshold,
    // [8K, 8K+64MiB) x_bf16, then 128 MiB w_bf16. (~201 MiB total)
    char* ws = (char*)d_ws;
    float* partials = (float*)ws;                            // 1024 floats
    float* thresh   = (float*)(ws + 4096);                   // 1 float
    unsigned short* x_bf = (unsigned short*)(ws + 8192);
    unsigned short* w_bf = x_bf + (size_t)M_TOT * IN_F;

    const int NW = OUT_F * IN_F;        // 67,108,864
    const int NX = M_TOT * IN_F;        // 33,554,432

    absum_partial<<<1024, 256, 0, stream>>>(w, partials, NW);
    absum_final  <<<1,    256, 0, stream>>>(partials, thresh, 1024, 1.0f / (float)NW);
    quant_w_kernel<<<2048, 256, 0, stream>>>((const float4*)w, thresh,
                                             (ushort4*)w_bf, NW / 4);
    cvt_x_kernel  <<<2048, 256, 0, stream>>>((const float4*)x,
                                             (ushort4*)x_bf, NX / 4);

    dim3 grid(OUT_F / BN, M_TOT / BM);  // (128, 64)
    bitlinear_wmma<<<grid, 256, 0, stream>>>(x_bf, w_bf, bias, out,
                                             M_TOT, OUT_F, IN_F);
}
